// MolecularGNN_47588237639681
// MI455X (gfx1250) — compile-verified, matrix operand order checked by reading, not run on hardware
//
#include <hip/hip_runtime.h>

// ---------------------------------------------------------------------------
// MolecularGNN forward for gfx1250 (MI455X): WMMA f16->f32 GEMMs for the
// message/update MLPs, async-to-LDS A-panel staging, atomics for segment_sum.
// ---------------------------------------------------------------------------

#define HIDC 256
constexpr int NNODES  = 10000;
constexpr int NEDGES  = 300000;
constexpr int NGRAPHS = 256;
constexpr int NODE_IN = 75;
constexpr int EDGE_IN = 12;
constexpr int NP = 10016;   // nodes padded to multiple of 32
constexpr int EP = 300032;  // edges padded to multiple of 32
constexpr float EPSLN = 1e-5f;
constexpr int ROWP = 776;   // LDS A-panel row stride in halves (768 + 8 pad)

typedef __attribute__((ext_vector_type(16))) _Float16 v16h;
typedef __attribute__((ext_vector_type(8)))  _Float16 v8h;
typedef __attribute__((ext_vector_type(8)))  float    v8f;

// Async global->LDS staging (gfx1250): issue GLOBAL_LOAD_ASYNC_TO_LDS_B128 via
// inline asm (GV mode: vdst = 32-bit LDS byte address, vaddr = 64-bit global
// address, saddr = off), track with ASYNCcnt via s_wait_asynccnt. Guarded on
// the wait builtin; falls back to plain load+store staging otherwise.
#if defined(__has_builtin)
#  if __has_builtin(__builtin_amdgcn_s_wait_asynccnt)
#    define HAVE_ASYNC_LDS 1
#  endif
#endif
#ifndef HAVE_ASYNC_LDS
#  define HAVE_ASYNC_LDS 0
#endif

// ---------------------------------------------------------------------------
// WMMA helpers (CDNA5 16x16x32 f16 -> f32)
// ---------------------------------------------------------------------------
__device__ __forceinline__ v8f wmma_f16(v16h a, v16h b, v8f c) {
  return __builtin_amdgcn_wmma_f32_16x16x32_f16(
      /*neg_a=*/false, a, /*neg_b=*/false, b,
      /*c_mod=*/(short)0, c, /*reuse_a=*/false, /*reuse_b=*/false);
}

// B fragments are pre-packed: 16 contiguous halves per lane per (ktile,ntile).
__device__ __forceinline__ v16h load_b_frag(const _Float16* __restrict__ pk,
                                            int ktile, int ntile, int NT, int lane) {
  const _Float16* p = pk + (((size_t)(ktile * NT + ntile) * 32 + lane) * 16);
  return *(const v16h*)p;
}

// A fragment from a row-major f16 row pointer (this lane's M row).
// 16-bit A 16x32 layout: lanes 0-15 hold K {kb..kb+7, kb+16..kb+23},
// lanes 16-31 hold K {kb+8..kb+15, kb+24..kb+31}.
__device__ __forceinline__ v16h make_a_frag(const _Float16* __restrict__ row,
                                            int kbase, int lane) {
  const int lg = (lane >> 4) & 1;
  const v8h lo = *(const v8h*)(row + kbase + lg * 8);
  const v8h hi = *(const v8h*)(row + kbase + 16 + lg * 8);
  v16h a;
#pragma unroll
  for (int i = 0; i < 8; i++) { a[i] = lo[i]; a[8 + i] = hi[i]; }
  return a;
}

// A fragment from LDS already stored in fragment layout (16 halves / lane).
__device__ __forceinline__ v16h load_a_lds(const _Float16* lds, int ks, int lane) {
  return *(const v16h*)(lds + ((ks * 32 + lane) * 16));
}

// Store one C-tile column (8 rows) with bias+ReLU into LDS in A-fragment layout.
__device__ __forceinline__ void store_hid_frag(_Float16* hidA, int hc, int mbase,
                                               const float* vals, float bias) {
  const int ks2 = hc >> 5, kr = hc & 31;
  int lg2, j;
  if      (kr <  8) { lg2 = 0; j = kr; }
  else if (kr < 16) { lg2 = 1; j = kr - 8; }
  else if (kr < 24) { lg2 = 0; j = 8 + (kr - 16); }
  else              { lg2 = 1; j = 8 + (kr - 24); }
#pragma unroll
  for (int r = 0; r < 8; r++) {
    float v = vals[r] + bias; v = v > 0.f ? v : 0.f;
    const int l2 = lg2 * 16 + (mbase + r);
    hidA[(ks2 * 32 + l2) * 16 + j] = (_Float16)v;
  }
}

__device__ __forceinline__ float wrsum(float v) {
#pragma unroll
  for (int o = 16; o > 0; o >>= 1) v += __shfl_xor(v, o, 32);
  return v;
}
__device__ __forceinline__ float wrmax(float v) {
#pragma unroll
  for (int o = 16; o > 0; o >>= 1) v = fmaxf(v, __shfl_xor(v, o, 32));
  return v;
}

// ---------------------------------------------------------------------------
// Weight repack: row-major f32 [K x N] -> f16 WMMA-fragment-major layout.
// ---------------------------------------------------------------------------
__global__ __launch_bounds__(256) void repack_kernel(const float* __restrict__ W,
                                                     _Float16* __restrict__ pk,
                                                     int K, int Nn) {
  const int idx = blockIdx.x * 256 + threadIdx.x;
  if (idx >= K * Nn) return;
  const int j  = idx & 15;
  const int l  = (idx >> 4) & 31;
  const int t  = idx >> 9;          // tile = ktile*NT + ntile
  const int NT = Nn >> 4;
  const int nt = t % NT, kt = t / NT;
  const int n  = nt * 16 + (l & 15);
  const int lg = l >> 4;
  const int kr = (j < 8) ? (lg * 8 + j) : (16 + lg * 8 + (j - 8));
  const int k  = kt * 32 + kr;
  pk[idx] = (_Float16)W[(size_t)k * Nn + n];
}

// ---------------------------------------------------------------------------
// Embeddings (small K, VALU)
// ---------------------------------------------------------------------------
__global__ __launch_bounds__(256) void node_emb_kernel(const float* __restrict__ nf,
    const float* __restrict__ w, const float* __restrict__ b,
    float* __restrict__ hcur, _Float16* __restrict__ hf) {
  const int n = blockIdx.x, c = threadIdx.x;
  __shared__ float sf[NODE_IN];
  if (c < NODE_IN) sf[c] = nf[(size_t)n * NODE_IN + c];
  __syncthreads();
  float acc = b[c];
#pragma unroll 5
  for (int k = 0; k < NODE_IN; k++) acc += sf[k] * w[k * HIDC + c];
  hcur[(size_t)n * HIDC + c] = acc;
  hf[(size_t)n * HIDC + c] = (_Float16)acc;
}

__global__ __launch_bounds__(256) void edge_emb_kernel(const float* __restrict__ efin,
    const float* __restrict__ w, const float* __restrict__ b,
    _Float16* __restrict__ ef) {
  const int e = blockIdx.x, c = threadIdx.x;
  __shared__ float sf[EDGE_IN];
  if (c < EDGE_IN) sf[c] = efin[(size_t)e * EDGE_IN + c];
  __syncthreads();
  float acc = b[c];
#pragma unroll
  for (int k = 0; k < EDGE_IN; k++) acc += sf[k] * w[k * HIDC + c];
  ef[(size_t)e * HIDC + c] = (_Float16)acc;
}

// ---------------------------------------------------------------------------
// Message MLP + scatter-add. One workgroup (8 waves) per 32-edge tile (M=32).
// Stage A-panel [32 x 768] f16 into LDS (async), then:
// GEMM1: [32x768]@[768x512] (+b,ReLU)  GEMM2: [32x512]@[512x256] (+b) -> agg
// ---------------------------------------------------------------------------
__global__ __launch_bounds__(256) void msg_kernel(
    const _Float16* __restrict__ hf, const _Float16* __restrict__ ef,
    const int* __restrict__ src, const int* __restrict__ tgt,
    const _Float16* __restrict__ w1p, const float* __restrict__ b1,
    const _Float16* __restrict__ w2p, const float* __restrict__ b2,
    float* __restrict__ agg) {
  __shared__ alignas(16) _Float16 hA[32 * ROWP];           // staged A panel
  __shared__ alignas(16) _Float16 hidA[2 * 16 * 32 * 16];  // GEMM2 A fragments
  __shared__ int ssrc[32];
  __shared__ int stgt[32];
  const int tile = blockIdx.x;
  const int tid  = threadIdx.x;
  const int lane = tid & 31;
  const int wave = tid >> 5;
  const int m    = lane & 15;

  if (tid < 32) {
    const int e  = tile * 32 + tid;
    const int ec = e < NEDGES ? e : (NEDGES - 1);
    ssrc[tid] = src[ec];
    stgt[tid] = tgt[ec];
  }
  __syncthreads();

  // Stage 32 rows x 768 halves of [h[src] | h[tgt] | e]: 3072 16B chunks.
#pragma unroll
  for (int it = 0; it < 12; it++) {
    const int q  = it * 256 + tid;
    const int mr = q / 96;           // row 0..31
    const int ck = q - mr * 96;      // 16B chunk 0..95 within row
    const int sec = ck >> 5;         // 0: h[src], 1: h[tgt], 2: e
    const int ko  = (ck & 31) * 8;   // half offset within 256-wide section
    const _Float16* gsrc;
    if (sec == 0) {
      gsrc = hf + (size_t)ssrc[mr] * HIDC + ko;
    } else if (sec == 1) {
      gsrc = hf + (size_t)stgt[mr] * HIDC + ko;
    } else {
      const int e  = tile * 32 + mr;
      const int ec = e < NEDGES ? e : (NEDGES - 1);
      gsrc = ef + (size_t)ec * HIDC + ko;
    }
    _Float16* ldst = hA + mr * ROWP + ck * 8;
#if HAVE_ASYNC_LDS
    // GV mode: LDS[vdst] = MEM[vaddr]; low 32 bits of the generic shared
    // pointer are the LDS byte offset (flat aperture truncation, ISA 10.2).
    const unsigned ldsoff = (unsigned)(size_t)(void*)ldst;
    const unsigned long long gaddr = (unsigned long long)(size_t)(const void*)gsrc;
    asm volatile("global_load_async_to_lds_b128 %0, %1, off"
                 :
                 : "v"(ldsoff), "v"(gaddr)
                 : "memory");
#else
    *(v8h*)ldst = *(const v8h*)gsrc;
#endif
  }
#if HAVE_ASYNC_LDS
  __builtin_amdgcn_s_wait_asynccnt(0);
#endif
  __syncthreads();

  const v8f z = {0.f, 0.f, 0.f, 0.f, 0.f, 0.f, 0.f, 0.f};

  // GEMM1: two M-tiles share every B fragment.
  v8f acc[2][4] = {{z, z, z, z}, {z, z, z, z}};
  const _Float16* row0 = hA + m * ROWP;
  const _Float16* row1 = hA + (16 + m) * ROWP;
  for (int ks = 0; ks < 24; ks++) {
    const int kb = ks * 32;
    const v16h a0 = make_a_frag(row0, kb, lane);
    const v16h a1 = make_a_frag(row1, kb, lane);
#pragma unroll
    for (int nt = 0; nt < 4; nt++) {
      const v16h b = load_b_frag(w1p, ks, wave * 4 + nt, 32, lane);
      acc[0][nt] = wmma_f16(a0, b, acc[0][nt]);
      acc[1][nt] = wmma_f16(a1, b, acc[1][nt]);
    }
  }

  // bias + ReLU -> LDS in GEMM2 A-fragment layout (per M-tile plane)
  {
    const int mb = (lane >> 4) * 8;
#pragma unroll
    for (int mt = 0; mt < 2; mt++) {
#pragma unroll
      for (int nt = 0; nt < 4; nt++) {
        const int hc = (wave * 4 + nt) * 16 + m;
        float tmp[8];
#pragma unroll
        for (int r = 0; r < 8; r++) tmp[r] = acc[mt][nt][r];
        store_hid_frag(hidA + mt * (16 * 32 * 16), hc, mb, tmp, b1[hc]);
      }
    }
  }
  __syncthreads();

  // GEMM2: two M-tiles share every B fragment.
  v8f d[2][2] = {{z, z}, {z, z}};
  for (int ks = 0; ks < 16; ks++) {
    const v16h a0 = load_a_lds(hidA, ks, lane);
    const v16h a1 = load_a_lds(hidA + 16 * 32 * 16, ks, lane);
#pragma unroll
    for (int nt = 0; nt < 2; nt++) {
      const v16h b = load_b_frag(w2p, ks, wave * 2 + nt, 16, lane);
      d[0][nt] = wmma_f16(a0, b, d[0][nt]);
      d[1][nt] = wmma_f16(a1, b, d[1][nt]);
    }
  }

  // scatter-add msg rows into agg[tgt]
  const int mb = (lane >> 4) * 8;
#pragma unroll
  for (int mt = 0; mt < 2; mt++) {
#pragma unroll
    for (int nt = 0; nt < 2; nt++) {
      const int c = (wave * 2 + nt) * 16 + m;
      const float bb = b2[c];
#pragma unroll
      for (int r = 0; r < 8; r++) {
        const int er = tile * 32 + mt * 16 + mb + r;
        if (er < NEDGES)
          atomicAdd(&agg[(size_t)stgt[mt * 16 + mb + r] * HIDC + c],
                    d[mt][nt][r] + bb);
      }
    }
  }
}

__global__ __launch_bounds__(256) void half_cast_kernel(const float* __restrict__ in,
                                                        _Float16* __restrict__ out) {
  const size_t i = (size_t)blockIdx.x * 256 + threadIdx.x;
  out[i] = (_Float16)in[i];
}

// ---------------------------------------------------------------------------
// Update MLP. One workgroup per 16-node tile.
// GEMM1: [16x512]@[512x256] (+b,ReLU)  GEMM2: [16x256]@[256x256] (+b) + residual
// ---------------------------------------------------------------------------
__global__ __launch_bounds__(256) void upd_kernel(
    const _Float16* __restrict__ hf, const _Float16* __restrict__ aggf,
    const _Float16* __restrict__ u1p, const float* __restrict__ ub1,
    const _Float16* __restrict__ u2p, const float* __restrict__ ub2,
    const float* __restrict__ hcur, float* __restrict__ hres) {
  __shared__ alignas(16) _Float16 hidA[8 * 32 * 16];  // 8 ksteps of K=256
  const int tile = blockIdx.x;
  const int lane = threadIdx.x & 31;
  const int wave = threadIdx.x >> 5;
  const int m = lane & 15;
  const int node = tile * 16 + m;
  const int nc = node < NNODES ? node : (NNODES - 1);
  const _Float16* rowH = hf + (size_t)nc * HIDC;
  const _Float16* rowA = aggf + (size_t)nc * HIDC;

  const v8f z = {0.f, 0.f, 0.f, 0.f, 0.f, 0.f, 0.f, 0.f};
  v8f acc[2] = {z, z};
#pragma unroll 2
  for (int ks = 0; ks < 16; ks++) {
    const int kb = ks * 32;
    const _Float16* row = (kb < 256) ? rowH : rowA;
    const v16h a = make_a_frag(row, kb & 255, lane);
#pragma unroll
    for (int nt = 0; nt < 2; nt++)
      acc[nt] = wmma_f16(a, load_b_frag(u1p, ks, wave * 2 + nt, 16, lane), acc[nt]);
  }
  {
    const int mb = (lane >> 4) * 8;
#pragma unroll
    for (int nt = 0; nt < 2; nt++) {
      const int hc = (wave * 2 + nt) * 16 + m;
      float tmp[8];
#pragma unroll
      for (int r = 0; r < 8; r++) tmp[r] = acc[nt][r];
      store_hid_frag(hidA, hc, mb, tmp, ub1[hc]);
    }
  }
  __syncthreads();

  v8f d[2] = {z, z};
  for (int ks = 0; ks < 8; ks++) {
    const v16h a = load_a_lds(hidA, ks, lane);
#pragma unroll
    for (int nt = 0; nt < 2; nt++)
      d[nt] = wmma_f16(a, load_b_frag(u2p, ks, wave * 2 + nt, 16, lane), d[nt]);
  }

  const int mb = (lane >> 4) * 8;
#pragma unroll
  for (int nt = 0; nt < 2; nt++) {
    const int c = (wave * 2 + nt) * 16 + m;
    const float bb = ub2[c];
#pragma unroll
    for (int r = 0; r < 8; r++) {
      const int nr = tile * 16 + mb + r;
      if (nr < NNODES)
        hres[(size_t)nr * HIDC + c] = hcur[(size_t)nr * HIDC + c] + d[nt][r] + bb;
    }
  }
}

// LayerNorm: one wave per row.
__global__ __launch_bounds__(256) void ln_kernel(const float* __restrict__ hres,
    const float* __restrict__ g, const float* __restrict__ bt,
    float* __restrict__ hcur, _Float16* __restrict__ hf) {
  const int row = blockIdx.x * 8 + (threadIdx.x >> 5);
  const int lane = threadIdx.x & 31;
  const float* p = hres + (size_t)row * HIDC;
  float v[8]; float s = 0.f;
#pragma unroll
  for (int i = 0; i < 8; i++) { v[i] = p[lane + i * 32]; s += v[i]; }
  const float mu = wrsum(s) * (1.f / 256.f);
  float q = 0.f;
#pragma unroll
  for (int i = 0; i < 8; i++) { const float dd = v[i] - mu; q += dd * dd; }
  const float inv = rsqrtf(wrsum(q) * (1.f / 256.f) + EPSLN);
#pragma unroll
  for (int i = 0; i < 8; i++) {
    const int c = lane + i * 32;
    const float o = (v[i] - mu) * inv * g[c] + bt[c];
    hcur[(size_t)row * HIDC + c] = o;
    hf[(size_t)row * HIDC + c] = (_Float16)o;
  }
}

// ---------------------------------------------------------------------------
// Attention logits, global softmax stats, weighted pooling, heads
// ---------------------------------------------------------------------------
__global__ __launch_bounds__(128) void att_kernel(const float* __restrict__ hcur,
    const float* __restrict__ w1, const float* __restrict__ b1,
    const float* __restrict__ w2, const float* __restrict__ b2,
    float* __restrict__ logits) {
  const int n = blockIdx.x, t = threadIdx.x;
  __shared__ float sh[HIDC];
  __shared__ float part[4];
  sh[t] = hcur[(size_t)n * HIDC + t];
  sh[t + 128] = hcur[(size_t)n * HIDC + t + 128];
  __syncthreads();
  float acc = b1[t];
#pragma unroll 8
  for (int k = 0; k < HIDC; k++) acc += sh[k] * w1[k * 128 + t];
  acc = acc > 0.f ? acc : 0.f;
  acc *= w2[t];
  const float s = wrsum(acc);
  if ((t & 31) == 0) part[t >> 5] = s;
  __syncthreads();
  if (t == 0) logits[n] = part[0] + part[1] + part[2] + part[3] + b2[0];
}

__global__ __launch_bounds__(1024) void softmax_stats_kernel(
    const float* __restrict__ logits, float* __restrict__ red) {
  __shared__ float part[32];
  __shared__ float bc;
  const int tid = threadIdx.x, lane = tid & 31, wv = tid >> 5;
  float mx = -3.4e38f;
  for (int i = tid; i < NNODES; i += 1024) mx = fmaxf(mx, logits[i]);
  mx = wrmax(mx);
  if (lane == 0) part[wv] = mx;
  __syncthreads();
  if (wv == 0) {
    float x = wrmax(part[lane]);
    if (lane == 0) bc = x;
  }
  __syncthreads();
  const float gm = bc;
  float s = 0.f;
  for (int i = tid; i < NNODES; i += 1024) s += expf(logits[i] - gm);
  s = wrsum(s);
  __syncthreads();
  if (lane == 0) part[wv] = s;
  __syncthreads();
  if (wv == 0) {
    float x = wrsum(part[lane]);
    if (lane == 0) { red[0] = gm; red[1] = x; }
  }
}

__global__ __launch_bounds__(256) void pool_kernel(const float* __restrict__ hcur,
    const float* __restrict__ logits, const float* __restrict__ red,
    const int* __restrict__ batch, float* __restrict__ graph) {
  const int n = blockIdx.x, c = threadIdx.x;
  const float w = expf(logits[n] - red[0]) / red[1];
  atomicAdd(&graph[(size_t)batch[n] * HIDC + c], hcur[(size_t)n * HIDC + c] * w);
}

__global__ __launch_bounds__(256) void heads_kernel(const float* __restrict__ graph,
    const float* __restrict__ pw, const float* __restrict__ pb,
    const float* __restrict__ fw, const float* __restrict__ fb,
    float* __restrict__ out_pred, float* __restrict__ out_feat) {
  const int g = blockIdx.x, c = threadIdx.x;
  __shared__ float sh[HIDC];
  __shared__ float part[8];
  sh[c] = graph[(size_t)g * HIDC + c];
  __syncthreads();
  float acc = fb[c];
#pragma unroll 8
  for (int k = 0; k < HIDC; k++) acc += sh[k] * fw[k * HIDC + c];
  out_feat[(size_t)g * HIDC + c] = acc;
  const float s = wrsum(sh[c] * pw[c]);
  if ((c & 31) == 0) part[c >> 5] = s;
  __syncthreads();
  if (c == 0) {
    float t = 0.f;
#pragma unroll
    for (int i = 0; i < 8; i++) t += part[i];
    out_pred[g] = t + pb[0];
  }
}

__global__ __launch_bounds__(256) void copy_kernel(const float* __restrict__ in,
                                                   float* __restrict__ out) {
  const size_t i = (size_t)blockIdx.x * 256 + threadIdx.x;
  out[i] = in[i];
}

// ---------------------------------------------------------------------------
// Launch: inputs follow setup_inputs() insertion order.
// ---------------------------------------------------------------------------
extern "C" void kernel_launch(void* const* d_in, const int* in_sizes, int n_in,
                              void* d_out, int out_size, void* d_ws, size_t ws_size,
                              hipStream_t stream) {
  (void)in_sizes; (void)n_in; (void)out_size; (void)ws_size;

  const float* node_features = (const float*)d_in[0];
  const float* edge_features = (const float*)d_in[1];
  const int*   edge_index    = (const int*)d_in[2];   // [2, E]: row0=src, row1=tgt
  const int*   batch_index   = (const int*)d_in[3];
  const float* node_emb_w = (const float*)d_in[4];
  const float* node_emb_b = (const float*)d_in[5];
  const float* edge_emb_w = (const float*)d_in[6];
  const float* edge_emb_b = (const float*)d_in[7];
  const float* att_w1 = (const float*)d_in[38];
  const float* att_b1 = (const float*)d_in[39];
  const float* att_w2 = (const float*)d_in[40];
  const float* att_b2 = (const float*)d_in[41];
  const float* prop_w = (const float*)d_in[42];
  const float* prop_b = (const float*)d_in[43];
  const float* feat_w = (const float*)d_in[44];
  const float* feat_b = (const float*)d_in[45];

  char* ws = (char*)d_ws;
  size_t off = 0;
  auto carve = [&](size_t bytes) -> void* {
    void* p = ws + off;
    off += (bytes + 255) & ~(size_t)255;
    return p;
  };
  float*    h_cur  = (float*)   carve((size_t)NP * HIDC * 4);
  _Float16* h_f16  = (_Float16*)carve((size_t)NP * HIDC * 2);
  _Float16* e_f16  = (_Float16*)carve((size_t)EP * HIDC * 2);
  float*    agg    = (float*)   carve((size_t)NP * HIDC * 4);
  _Float16* agg16  = (_Float16*)carve((size_t)NP * HIDC * 2);
  float*    hres   = (float*)   carve((size_t)NP * HIDC * 4);
  float*    graph  = (float*)   carve((size_t)NGRAPHS * HIDC * 4);
  float*    logits = (float*)   carve((size_t)NP * 4);
  float*    red    = (float*)   carve(256);
  _Float16 *w1p[3], *w2p[3], *u1p[3], *u2p[3];
  for (int l = 0; l < 3; l++) {
    w1p[l] = (_Float16*)carve((size_t)768 * 512 * 2);
    w2p[l] = (_Float16*)carve((size_t)512 * 256 * 2);
    u1p[l] = (_Float16*)carve((size_t)512 * 256 * 2);
    u2p[l] = (_Float16*)carve((size_t)256 * 256 * 2);
  }

  // Repack all layer weights into WMMA fragment layout (f16).
  for (int l = 0; l < 3; l++) {
    const int base = 8 + 10 * l;
    repack_kernel<<<(768 * 512 + 255) / 256, 256, 0, stream>>>(
        (const float*)d_in[base + 0], w1p[l], 768, 512);
    repack_kernel<<<(512 * 256 + 255) / 256, 256, 0, stream>>>(
        (const float*)d_in[base + 2], w2p[l], 512, 256);
    repack_kernel<<<(512 * 256 + 255) / 256, 256, 0, stream>>>(
        (const float*)d_in[base + 4], u1p[l], 512, 256);
    repack_kernel<<<(256 * 256 + 255) / 256, 256, 0, stream>>>(
        (const float*)d_in[base + 6], u2p[l], 256, 256);
  }

  node_emb_kernel<<<NNODES, 256, 0, stream>>>(node_features, node_emb_w, node_emb_b,
                                              h_cur, h_f16);
  edge_emb_kernel<<<NEDGES, 256, 0, stream>>>(edge_features, edge_emb_w, edge_emb_b,
                                              e_f16);

  const int* src = edge_index;
  const int* tgt = edge_index + NEDGES;
  for (int l = 0; l < 3; l++) {
    const int base = 8 + 10 * l;
    (void)hipMemsetAsync(agg, 0, (size_t)NP * HIDC * 4, stream);
    msg_kernel<<<EP / 32, 256, 0, stream>>>(
        h_f16, e_f16, src, tgt,
        w1p[l], (const float*)d_in[base + 1],
        w2p[l], (const float*)d_in[base + 3], agg);
    half_cast_kernel<<<(NP * HIDC) / 256, 256, 0, stream>>>(agg, agg16);
    upd_kernel<<<NP / 16, 256, 0, stream>>>(
        h_f16, agg16,
        u1p[l], (const float*)d_in[base + 5],
        u2p[l], (const float*)d_in[base + 7], h_cur, hres);
    ln_kernel<<<NP / 8, 256, 0, stream>>>(
        hres, (const float*)d_in[base + 8], (const float*)d_in[base + 9],
        h_cur, h_f16);
  }

  att_kernel<<<NNODES, 128, 0, stream>>>(h_cur, att_w1, att_b1, att_w2, att_b2, logits);
  softmax_stats_kernel<<<1, 1024, 0, stream>>>(logits, red);
  (void)hipMemsetAsync(graph, 0, (size_t)NGRAPHS * HIDC * 4, stream);
  pool_kernel<<<NNODES, 256, 0, stream>>>(h_cur, logits, red, batch_index, graph);

  float* out_pred = (float*)d_out;                 // [256]
  float* out_feat = out_pred + NGRAPHS;            // [256,256]
  float* out_h    = out_feat + NGRAPHS * HIDC;     // [10000,256]
  heads_kernel<<<NGRAPHS, 256, 0, stream>>>(graph, prop_w, prop_b, feat_w, feat_b,
                                            out_pred, out_feat);
  copy_kernel<<<(NNODES * HIDC) / 256, 256, 0, stream>>>(h_cur, out_h);
}
